// ORG_41515153883458
// MI455X (gfx1250) — matrix operand order repaired and checked by reading, not compiled, not run on hardware
//
#include <hip/hip_runtime.h>

// ---------------------------------------------------------------------------
// Problem constants (reference: B=32, F=32, N=36, D=2048, P=512)
// ---------------------------------------------------------------------------
#define NOBJ   36
#define DDIM   2048
#define PDIM   512
#define MROWS  36864            // B*F*N
#define BFCNT  1024             // B*F

typedef __attribute__((ext_vector_type(16))) __bf16        v16bf;
typedef __attribute__((ext_vector_type(8)))  float         v8f;
typedef __attribute__((ext_vector_type(4)))  unsigned int  u32x4;
typedef __attribute__((ext_vector_type(4)))  float         f32x4;

union Frag { v16bf v; u32x4 q[2]; };

// fp32 -> bf16 bits, round-to-nearest-even
static __device__ __forceinline__ unsigned short f2bf_bits(float f) {
    unsigned int u = __builtin_bit_cast(unsigned int, f);
    u += 0x7FFFu + ((u >> 16) & 1u);
    return (unsigned short)(u >> 16);
}
static __device__ __forceinline__ __bf16 bits2bf(unsigned short b) {
    return __builtin_bit_cast(__bf16, b);
}
static __device__ __forceinline__ unsigned int pack2(float lo, float hi) {
    return (unsigned int)f2bf_bits(lo) | ((unsigned int)f2bf_bits(hi) << 16);
}

// CDNA5 async DMA: copy 16 bytes global -> LDS per lane, tracked by ASYNCcnt.
static __device__ __forceinline__ void async_b128(unsigned lds_off, const void* gaddr) {
    asm volatile("global_load_async_to_lds_b128 %0, %1, off"
                 :: "v"(lds_off), "v"(gaddr) : "memory");
}

// ---------------------------------------------------------------------------
// Kernel 0a: x fp32 -> bf16 (vectorized: 2x b128 load -> 1x b128 store)
// ---------------------------------------------------------------------------
__global__ __launch_bounds__(256) void cvt_x_bf16(const float* __restrict__ in,
                                                  unsigned short* __restrict__ out) {
    const size_t i = ((size_t)blockIdx.x * 256 + threadIdx.x) * 8;
    f32x4 a = *(const f32x4*)(in + i);
    f32x4 b = *(const f32x4*)(in + i + 4);
    u32x4 p;
    p[0] = pack2(a[0], a[1]);
    p[1] = pack2(a[2], a[3]);
    p[2] = pack2(b[0], b[1]);
    p[3] = pack2(b[2], b[3]);
    *(u32x4*)(out + i) = p;
}

// ---------------------------------------------------------------------------
// Kernel 0b: W [K=2048][N=512] fp32 -> WT [4][N=512][K=2048] bf16 (LDS-tiled
// transpose so both global read and write are coalesced)
// ---------------------------------------------------------------------------
__global__ __launch_bounds__(256) void cvt_w_bf16t(
    const float* __restrict__ W0, const float* __restrict__ W1,
    const float* __restrict__ W2, const float* __restrict__ W3,
    unsigned short* __restrict__ WT)
{
    __shared__ unsigned short T[32][33];
    const int k0 = blockIdx.x * 32;
    const int n0 = blockIdx.y * 32;
    const int w  = blockIdx.z;
    const float* W = (w == 0) ? W0 : (w == 1) ? W1 : (w == 2) ? W2 : W3;
    unsigned short* out = WT + (size_t)w * PDIM * DDIM;

    const int tx = threadIdx.x & 31, ty = threadIdx.x >> 5;
    #pragma unroll
    for (int r = ty; r < 32; r += 8)
        T[r][tx] = f2bf_bits(W[(size_t)(k0 + r) * PDIM + (n0 + tx)]);
    __syncthreads();
    #pragma unroll
    for (int r = ty; r < 32; r += 8)
        out[(size_t)(n0 + r) * DDIM + (k0 + tx)] = T[tx][r];
}

// ---------------------------------------------------------------------------
// Kernel 1: fused 4-way projection GEMM, bf16 WMMA.
//   Staging = pure async-to-LDS DMA (ASYNCcnt), double-buffered LDS,
//   DMA of tile t+1 overlaps WMMA of tile t.
// Block tile 64M x 128N, 8 waves (2x4), each wave 32x32 (4 accum tiles).
// ---------------------------------------------------------------------------
__global__ __launch_bounds__(256) void proj_gemm4(
    const unsigned short* __restrict__ xbf,   // [36864][2048] bf16
    const unsigned short* __restrict__ WT,    // [4][512][2048] bf16 (pre-transposed)
    const float* __restrict__ bp, const float* __restrict__ bsg,
    const float* __restrict__ bps,
    float* __restrict__ r_feat,
    unsigned short* __restrict__ sig,
    unsigned short* __restrict__ psi,
    unsigned short* __restrict__ vmat)
{
    __shared__ __align__(16) unsigned short As[2][64][40];    // [m][k] bf16 bits
    __shared__ __align__(16) unsigned short Bs[2][128][40];   // [n][k] bf16 bits
    const unsigned ABYTES = 64 * 40 * 2;    // 5120
    const unsigned BBYTES = 128 * 40 * 2;   // 10240

    const int tid   = threadIdx.x;
    const int lane  = tid & 31;
    const int wave  = tid >> 5;
    const int wy    = wave >> 2;       // 0..1
    const int wx    = wave & 3;        // 0..3
    const int m0    = blockIdx.x * 64;
    const int n0    = blockIdx.y * 128;
    const int which = blockIdx.z;      // 0:proj 1:sigma 2:psi 3:r

    const unsigned short* Wt = WT + (size_t)which * PDIM * DDIM;

    const unsigned as0 = (unsigned)(size_t)(const void*)&As[0][0][0];
    const unsigned bs0 = (unsigned)(size_t)(const void*)&Bs[0][0][0];

    // Per-thread DMA chunk assignment (16B per lane)
    const int ar = tid >> 2;             // A row 0..63
    const int ac = (tid & 3) << 3;       // A k-col 0,8,16,24
    const unsigned a_lds = (unsigned)((ar * 40 + ac) * 2);
    const unsigned short* a_g = xbf + (size_t)(m0 + ar) * DDIM + ac;

    v8f acc[4];
    #pragma unroll
    for (int t = 0; t < 4; ++t)
        #pragma unroll
        for (int i = 0; i < 8; ++i) acc[t][i] = 0.0f;

    const int rl  = lane & 15;
    const int akb = (lane < 16) ? 0 : 16;   // A frag: 16 contiguous K
    const int bkb = (lane < 16) ? 0 : 16;   // B frag: 16 contiguous K

    // Prologue: DMA tile 0 into buffer 0
    {
        async_b128(as0 + a_lds, a_g);
        #pragma unroll
        for (int j = 0; j < 2; ++j) {
            const int idx = tid + j * 256;
            const int br = idx >> 2, bc = (idx & 3) << 3;
            async_b128(bs0 + (unsigned)((br * 40 + bc) * 2),
                       Wt + (size_t)(n0 + br) * DDIM + bc);
        }
    }

    for (int k0 = 0; k0 < DDIM; k0 += 32) {
        const int buf = (k0 >> 5) & 1;
        if (k0 + 32 < DDIM) {
            // Issue DMA for tile t+1 into the other buffer, then wait for tile t.
            const int kn = k0 + 32;
            async_b128(as0 + (buf ^ 1) * ABYTES + a_lds, a_g + kn);
            #pragma unroll
            for (int j = 0; j < 2; ++j) {
                const int idx = tid + j * 256;
                const int br = idx >> 2, bc = (idx & 3) << 3;
                async_b128(bs0 + (buf ^ 1) * BBYTES + (unsigned)((br * 40 + bc) * 2),
                           Wt + (size_t)(n0 + br) * DDIM + kn + bc);
            }
            asm volatile("s_wait_asynccnt 3" ::: "memory");
        } else {
            asm volatile("s_wait_asynccnt 0" ::: "memory");
        }
        __syncthreads();

        Frag fa[2], fb[2];
        #pragma unroll
        for (int mt = 0; mt < 2; ++mt) {
            const unsigned short* ap = &As[buf][wy * 32 + mt * 16 + rl][0];
            fa[mt].q[0] = *(const u32x4*)(ap + akb);
            fa[mt].q[1] = *(const u32x4*)(ap + akb + 8);
        }
        #pragma unroll
        for (int nt = 0; nt < 2; ++nt) {
            const unsigned short* bpr = &Bs[buf][wx * 32 + nt * 16 + rl][0];
            fb[nt].q[0] = *(const u32x4*)(bpr + bkb);
            fb[nt].q[1] = *(const u32x4*)(bpr + bkb + 8);
        }
        #pragma unroll
        for (int mt = 0; mt < 2; ++mt)
            #pragma unroll
            for (int nt = 0; nt < 2; ++nt)
                acc[mt * 2 + nt] = __builtin_amdgcn_wmma_f32_16x16x32_bf16(
                    false, fa[mt].v, false, fb[nt].v,
                    (short)0, acc[mt * 2 + nt], false, false);
        __syncthreads();
    }

    // Epilogue: bias + store (C layout: lane<16 -> M=v, lane>=16 -> M=v+8; N=lane%16)
    const int rbase = (lane < 16) ? 0 : 8;
    #pragma unroll
    for (int mt = 0; mt < 2; ++mt) {
        #pragma unroll
        for (int nt = 0; nt < 2; ++nt) {
            const int col = n0 + wx * 32 + nt * 16 + rl;
            float bias = 0.0f;
            if      (which == 0) bias = bp[col];
            else if (which == 1) bias = bsg[col];
            else if (which == 2) bias = bps[col];
            #pragma unroll
            for (int v = 0; v < 8; ++v) {
                const int row = m0 + wy * 32 + mt * 16 + rbase + v;
                const size_t idx = (size_t)row * PDIM + col;
                const float val = acc[mt * 2 + nt][v] + bias;
                if      (which == 0) r_feat[idx] = val;
                else if (which == 1) sig[idx]  = f2bf_bits(val);
                else if (which == 2) psi[idx]  = f2bf_bits(val);
                else                 vmat[idx] = f2bf_bits(val);
            }
        }
    }
}

// ---------------------------------------------------------------------------
// Kernel 2: per-(b,f) attention: S = sig psi^T (36x36), softmax, r_hat = A v
// 288 threads = 9 waves. Scores: 3x3 WMMA tiles (pad 36->48). AV: 3x32 tiles.
// ---------------------------------------------------------------------------
__global__ __launch_bounds__(288) void attention_bf(
    const unsigned short* __restrict__ sig,
    const unsigned short* __restrict__ psi,
    const unsigned short* __restrict__ vmat,
    float* __restrict__ rhat)
{
    __shared__ float S[48][49];
    __shared__ __align__(16) unsigned short Ah[48][64];   // a_hat bf16, K-padded to 64

    const int bf   = blockIdx.x;
    const int tid  = threadIdx.x;
    const int lane = tid & 31;
    const int wave = tid >> 5;          // 0..8
    const int rl   = lane & 15;
    const int akb  = (lane < 16) ? 0 : 16;
    const int bkb  = (lane < 16) ? 0 : 16;
    const int rbase = (lane < 16) ? 0 : 8;
    const size_t base = (size_t)bf * NOBJ * PDIM;

    const u32x4 zq = {0u, 0u, 0u, 0u};

    // ---- Phase 1: scores S = sig . psi^T  (K = 512) ----
    {
        const int ti = wave / 3, tj = wave % 3;
        v8f acc;
        #pragma unroll
        for (int i = 0; i < 8; ++i) acc[i] = 0.0f;

        const int arow = ti * 16 + rl;
        const int bcol = tj * 16 + rl;
        const bool aval = (arow < NOBJ);
        const bool bval = (bcol < NOBJ);
        const unsigned short* ap = sig + base + (size_t)arow * PDIM;
        const unsigned short* bp = psi + base + (size_t)bcol * PDIM;

        for (int k0 = 0; k0 < PDIM; k0 += 32) {
            Frag fa, fb;
            if (aval) {
                fa.q[0] = *(const u32x4*)(ap + k0 + akb);
                fa.q[1] = *(const u32x4*)(ap + k0 + akb + 8);
            } else { fa.q[0] = zq; fa.q[1] = zq; }
            if (bval) {
                fb.q[0] = *(const u32x4*)(bp + k0 + bkb);
                fb.q[1] = *(const u32x4*)(bp + k0 + bkb + 8);
            } else { fb.q[0] = zq; fb.q[1] = zq; }
            acc = __builtin_amdgcn_wmma_f32_16x16x32_bf16(
                false, fa.v, false, fb.v, (short)0, acc, false, false);
        }
        const int srow = ti * 16 + rbase;
        const int scol = tj * 16 + rl;
        #pragma unroll
        for (int v = 0; v < 8; ++v) S[srow + v][scol] = acc[v];
    }
    __syncthreads();

    // ---- Phase 2: row softmax over valid 36x36, write a_hat (bf16, padded) ----
    if (tid < 48) {
        const int row = tid;
        if (row < NOBJ) {
            float mx = -3.4e38f;
            for (int m = 0; m < NOBJ; ++m) mx = fmaxf(mx, S[row][m]);
            float sum = 0.0f;
            for (int m = 0; m < NOBJ; ++m) {
                float e = __expf(S[row][m] - mx);
                S[row][m] = e;
                sum += e;
            }
            const float inv = 1.0f / sum;
            for (int m = 0; m < NOBJ; ++m) Ah[row][m] = f2bf_bits(S[row][m] * inv);
            for (int m = NOBJ; m < 64; ++m) Ah[row][m] = 0;
        } else {
            for (int m = 0; m < 64; ++m) Ah[row][m] = 0;
        }
    }
    __syncthreads();

    // ---- Phase 3: r_hat = a_hat . v   (M=48 pad, N=512, K=64 pad) ----
    for (int t = wave; t < 3 * 32; t += 9) {
        const int ti = t >> 5;          // 0..2
        const int tj = t & 31;          // 0..31
        v8f acc;
        #pragma unroll
        for (int i = 0; i < 8; ++i) acc[i] = 0.0f;

        const int arow = ti * 16 + rl;                 // < 48, always valid in Ah
        const int col  = tj * 16 + rl;                 // < 512
        #pragma unroll
        for (int k0 = 0; k0 < 64; k0 += 32) {
            Frag fa, fb;
            const unsigned short* ap = &Ah[arow][0];
            fa.q[0] = *(const u32x4*)(ap + k0 + akb);
            fa.q[1] = *(const u32x4*)(ap + k0 + akb + 8);
            // V fragment: element-predicated (m >= 36 contributes exact zero)
            #pragma unroll
            for (int j = 0; j < 16; ++j) {
                const int m = k0 + bkb + j;
                fb.v[j] = (m < NOBJ)
                    ? bits2bf(vmat[base + (size_t)m * PDIM + col])
                    : bits2bf((unsigned short)0);
            }
            acc = __builtin_amdgcn_wmma_f32_16x16x32_bf16(
                false, fa.v, false, fb.v, (short)0, acc, false, false);
        }
        #pragma unroll
        for (int v = 0; v < 8; ++v) {
            const int row = ti * 16 + rbase + v;
            if (row < NOBJ)
                rhat[base + (size_t)row * PDIM + col] = acc[v];
        }
    }
}

// ---------------------------------------------------------------------------
// Launch
// ---------------------------------------------------------------------------
extern "C" void kernel_launch(void* const* d_in, const int* in_sizes, int n_in,
                              void* d_out, int out_size, void* d_ws, size_t ws_size,
                              hipStream_t stream) {
    const float* x   = (const float*)d_in[0];
    const float* Wp  = (const float*)d_in[1];
    const float* bp  = (const float*)d_in[2];
    const float* Wsg = (const float*)d_in[3];
    const float* bsg = (const float*)d_in[4];
    const float* Wps = (const float*)d_in[5];
    const float* bps = (const float*)d_in[6];
    const float* Wr  = (const float*)d_in[7];

    const size_t MP = (size_t)MROWS * PDIM;      // 18,874,368 elements
    float* r_feat = (float*)d_out;
    float* r_hat  = (float*)d_out + MP;

    // Workspace (bf16 elements):
    //   vmat first (padded tail reads in attention stay in-bounds),
    //   then sig, psi, x-bf16, WT-bf16.  Total ~273 MB.
    unsigned short* vmat = (unsigned short*)d_ws;
    unsigned short* sig  = vmat + MP;
    unsigned short* psi  = sig + MP;
    unsigned short* xbf  = psi + MP;                       // [36864][2048]
    unsigned short* WT   = xbf + (size_t)MROWS * DDIM;     // [4][512][2048]

    // Pre-convert x (once) and W (transposed, once)
    cvt_x_bf16<<<dim3((unsigned)((size_t)MROWS * DDIM / (256 * 8))), 256, 0, stream>>>(x, xbf);
    cvt_w_bf16t<<<dim3(DDIM / 32, PDIM / 32, 4), 256, 0, stream>>>(Wp, Wsg, Wps, Wr, WT);

    // Projections: grid (M/64, P/128, 4 weights)
    proj_gemm4<<<dim3(MROWS / 64, PDIM / 128, 4), 256, 0, stream>>>(
        xbf, WT, bp, bsg, bps, r_feat, sig, psi, vmat);

    // Attention: one block per (b,f)
    attention_bf<<<dim3(BFCNT), 288, 0, stream>>>(sig, psi, vmat, r_hat);
}
